// MultiHeadSelfAttention_42640435314766
// MI455X (gfx1250) — matrix operand order repaired and checked by reading, not compiled, not run on hardware
//
#include <hip/hip_runtime.h>
#include <hip/hip_bf16.h>

typedef __attribute__((ext_vector_type(16))) _Float16 v16h;
typedef __attribute__((ext_vector_type(8)))  float    v8f;
typedef __attribute__((address_space(3))) _Float16 lds_f16_t;

#define WMMA16(A,B,C) __builtin_amdgcn_wmma_f32_16x16x32_f16(false,(A),false,(B),(short)0,(C),false,false)

// A-fragment (16xK tile, K chunk of 32) from row-major [row][k] f16 storage.
// Layout: lane m = lane&15, hk = lane>>4; elems 0..15 -> K = hk*16+{0..15}.
static __device__ inline v16h load_fragA(const _Float16* base, int ld, int row0, int k0, int lane) {
  const int r  = lane & 15;
  const int hk = lane >> 4;
  const _Float16* p = base + (size_t)(row0 + r) * ld + (k0 + hk * 16);
  v16h out;
  ((float4*)&out)[0] = *(const float4*)(p);
  ((float4*)&out)[1] = *(const float4*)(p + 8);
  return out;
}

// B-fragment (Kx16 tile, K chunk of 32) where storage is row-major [n][k]:
// lane n = lane&15 holds 16 contiguous K values starting at (lane>>4)*16.
static __device__ inline v16h load_fragB(const _Float16* base, int ld, int col0, int k0, int lane) {
  const _Float16* p = base + (size_t)(col0 + (lane & 15)) * ld + (k0 + (lane >> 4) * 16);
  v16h out;
  ((float4*)&out)[0] = *(const float4*)(p);
  ((float4*)&out)[1] = *(const float4*)(p + 8);
  return out;
}

// Async DMA of 16 bytes global -> LDS (per lane), tracked by ASYNCcnt.
static __device__ inline void async_copy16(const _Float16* g, _Float16* l) {
  const uint32_t loff = (uint32_t)(uintptr_t)(lds_f16_t*)l;      // LDS byte offset
  const unsigned long long ga = (unsigned long long)(uintptr_t)g;
  asm volatile("global_load_async_to_lds_b128 %0, %1, off"
               :: "v"(loff), "v"(ga) : "memory");
}
static __device__ inline void wait_async0() {
  asm volatile("s_wait_asynccnt 0" ::: "memory");
}

__global__ void cvt_f16_kernel(const float* __restrict__ s, _Float16* __restrict__ d, int n) {
  int i = blockIdx.x * blockDim.x + threadIdx.x;
  if (i < n) d[i] = (_Float16)s[i];
}

__global__ void zero_kernel(uint4* __restrict__ d, int n4) {
  int i = blockIdx.x * blockDim.x + threadIdx.x;
  if (i < n4) d[i] = make_uint4(0u, 0u, 0u, 0u);
}

// One block (384 threads) per token: LayerNorm over C=384, emit f16 tokens.
__global__ __launch_bounds__(384) void ln_kernel(const float* __restrict__ x,
                                                 const float* __restrict__ lw,
                                                 const float* __restrict__ lb,
                                                 _Float16* __restrict__ t) {
  const int token = blockIdx.x;           // 0..8191
  const int b = token >> 10, n = token & 1023;
  const int c = threadIdx.x;              // 0..383
  const float v = x[((size_t)(b * 384 + c)) * 1024 + n];

  __shared__ float part[12];
  __shared__ float stat[2];

  float s = v;
  #pragma unroll
  for (int m = 16; m >= 1; m >>= 1) s += __shfl_xor(s, m, 32);
  if ((threadIdx.x & 31) == 0) part[threadIdx.x >> 5] = s;
  __syncthreads();
  if (threadIdx.x < 32) {
    float tot = (threadIdx.x < 12) ? part[threadIdx.x] : 0.0f;
    #pragma unroll
    for (int m = 8; m >= 1; m >>= 1) tot += __shfl_xor(tot, m, 32);
    if (threadIdx.x == 0) stat[0] = tot * (1.0f / 384.0f);
  }
  __syncthreads();
  const float mu = stat[0];
  const float dv = v - mu;

  float s2 = dv * dv;
  #pragma unroll
  for (int m = 16; m >= 1; m >>= 1) s2 += __shfl_xor(s2, m, 32);
  __syncthreads();
  if ((threadIdx.x & 31) == 0) part[threadIdx.x >> 5] = s2;
  __syncthreads();
  if (threadIdx.x < 32) {
    float tot = (threadIdx.x < 12) ? part[threadIdx.x] : 0.0f;
    #pragma unroll
    for (int m = 8; m >= 1; m >>= 1) tot += __shfl_xor(tot, m, 32);
    if (threadIdx.x == 0) stat[1] = rsqrtf(tot * (1.0f / 384.0f) + 1e-6f);
  }
  __syncthreads();
  t[(size_t)token * 384 + c] = (_Float16)(dv * stat[1] * lw[c] + lb[c]);
}

// QKV GEMM: [8192,384]f16 x Wqkv^T[384,1152] -> scatter into head-major
// Q[bh][n][64], K[bh][n][64] (d padded to 64), Vt[bh][64][n].
// K-loop is register double-buffered: next step's fragments are in flight
// while the current step's WMMAs execute.
__global__ __launch_bounds__(256) void qkv_gemm_kernel(const _Float16* __restrict__ t,
                                                       const _Float16* __restrict__ w,
                                                       _Float16* __restrict__ Qb,
                                                       _Float16* __restrict__ Kb,
                                                       _Float16* __restrict__ Vt) {
  const int wid  = (blockIdx.x * 256 + (int)threadIdx.x) >> 5;   // 0..9215
  const int lane = threadIdx.x & 31;
  const int mt = wid & 511;    // token tile (16 rows)
  const int jg = wid >> 9;     // 0..17, 64 output cols each
  v8f a0 = {}, a1 = {}, a2 = {}, a3 = {};

  v16h af = load_fragA(t, 384, mt * 16, 0, lane);
  v16h b0 = load_fragB(w, 384, jg * 64 +  0, 0, lane);
  v16h b1 = load_fragB(w, 384, jg * 64 + 16, 0, lane);
  v16h b2 = load_fragB(w, 384, jg * 64 + 32, 0, lane);
  v16h b3 = load_fragB(w, 384, jg * 64 + 48, 0, lane);
  for (int kb = 0; kb < 384 - 32; kb += 32) {
    const int kn = kb + 32;
    const v16h afn = load_fragA(t, 384, mt * 16, kn, lane);
    const v16h b0n = load_fragB(w, 384, jg * 64 +  0, kn, lane);
    const v16h b1n = load_fragB(w, 384, jg * 64 + 16, kn, lane);
    const v16h b2n = load_fragB(w, 384, jg * 64 + 32, kn, lane);
    const v16h b3n = load_fragB(w, 384, jg * 64 + 48, kn, lane);
    a0 = WMMA16(af, b0, a0);
    a1 = WMMA16(af, b1, a1);
    a2 = WMMA16(af, b2, a2);
    a3 = WMMA16(af, b3, a3);
    af = afn; b0 = b0n; b1 = b1n; b2 = b2n; b3 = b3n;
  }
  a0 = WMMA16(af, b0, a0);
  a1 = WMMA16(af, b1, a1);
  a2 = WMMA16(af, b2, a2);
  a3 = WMMA16(af, b3, a3);

  const int g = lane >> 4, col = lane & 15;
  v8f accs[4] = {a0, a1, a2, a3};
  #pragma unroll
  for (int jj = 0; jj < 4; ++jj) {
    const int jt = jg * 4 + jj;                 // 16-col tile, 0..71
    const int s = jt / 24, rem = jt % 24;       // s: 0=Q 1=K 2=V
    const int head = rem / 3;
    const int dd   = (rem % 3) * 16 + col;      // 0..47
    #pragma unroll
    for (int r = 0; r < 8; ++r) {
      const int token = mt * 16 + r + 8 * g;
      const int b = token >> 10, n = token & 1023;
      const int bh = b * 8 + head;
      const _Float16 val = (_Float16)accs[jj][r];
      if (s == 0)      Qb[((size_t)bh * 1024 + n) * 64 + dd] = val;
      else if (s == 1) Kb[((size_t)bh * 1024 + n) * 64 + dd] = val;
      else             Vt[((size_t)bh * 64 + dd) * 1024 + n] = val;
    }
  }
}

// Flash attention: block = 8 waves, each wave owns 16 query rows of one (b,h).
// K/V tiles (32 keys) are staged once per block into LDS with async-to-LDS
// DMA, double-buffered so tile i+1's DMA overlaps tile i's WMMA + softmax.
__global__ __launch_bounds__(256) void attn_kernel(const _Float16* __restrict__ Qb,
                                                   const _Float16* __restrict__ Kb,
                                                   const _Float16* __restrict__ Vtg,
                                                   _Float16* __restrict__ ao) {
  __shared__ _Float16 Plds[8][16][32];   // per-wave P bounce (C-layout -> A-frag)
  __shared__ _Float16 Kt[2][32][64];     // double-buffered K tile: 32 keys x 64 d
  __shared__ _Float16 Vts[2][64][32];    // double-buffered V tile: 64 d x 32 keys

  const int tid = threadIdx.x;
  const int w = tid >> 5, lane = tid & 31;
  const int bh = blockIdx.x >> 3, qblk = blockIdx.x & 7;
  const int b = bh >> 3, h = bh & 7;
  const _Float16* Q = Qb  + (size_t)bh * 1024 * 64;
  const _Float16* K = Kb  + (size_t)bh * 1024 * 64;
  const _Float16* V = Vtg + (size_t)bh * 64 * 1024;
  const int q0 = qblk * 128 + w * 16;
  const int g = lane >> 4, col = lane & 15;

  const v16h qa0 = load_fragA(Q, 64, q0, 0, lane);   // d 0..31
  const v16h qa1 = load_fragA(Q, 64, q0, 32, lane);  // d 32..63 (zero-padded)

  // K-tile stage: 32 rows x 128B = 256 chunks of 16B; V-tile: 64 rows x 64B.
  const int krow = tid >> 3, kch = (tid & 7) * 8;    // K: row 0..31, chunk 0..7
  const int vrow = tid >> 2, vch = (tid & 3) * 8;    // V: row 0..63, chunk 0..3

  // stage tile 0
  async_copy16(K + (size_t)(0 + krow) * 64 + kch, &Kt[0][krow][kch]);
  async_copy16(V + (size_t)vrow * 1024 + 0 + vch, &Vts[0][vrow][vch]);
  wait_async0();
  __syncthreads();

  float mst[8], lst[8];
  #pragma unroll
  for (int r = 0; r < 8; ++r) { mst[r] = -1e30f; lst[r] = 0.0f; }
  v8f o0 = {}, o1 = {}, o2 = {};
  const float sc = 0.14433756729740643f;             // 48^-0.5

  for (int it = 0; it < 32; ++it) {
    const int cur = it & 1;
    if (it + 1 < 32) {                               // prefetch tile it+1
      const int kn = (it + 1) * 32;
      async_copy16(K + (size_t)(kn + krow) * 64 + kch, &Kt[1 - cur][krow][kch]);
      async_copy16(V + (size_t)vrow * 1024 + kn + vch, &Vts[1 - cur][vrow][vch]);
    }

    const _Float16* kbase = &Kt[cur][0][0];
    v8f s0 = {}, s1 = {};
    {
      const v16h kb0 = load_fragB(kbase, 64, 0,  0,  lane);
      const v16h kb1 = load_fragB(kbase, 64, 0,  32, lane);
      s0 = WMMA16(qa0, kb0, s0);
      s0 = WMMA16(qa1, kb1, s0);
      const v16h kb2 = load_fragB(kbase, 64, 16, 0,  lane);
      const v16h kb3 = load_fragB(kbase, 64, 16, 32, lane);
      s1 = WMMA16(qa0, kb2, s1);
      s1 = WMMA16(qa1, kb3, s1);
    }
    float alpha[8];
    #pragma unroll
    for (int r = 0; r < 8; ++r) {
      const float e0 = s0[r] * sc;
      const float e1 = s1[r] * sc;
      s0[r] = e0; s1[r] = e1;
      float mx = fmaxf(e0, e1);
      mx = fmaxf(mx, __shfl_xor(mx, 1, 32));
      mx = fmaxf(mx, __shfl_xor(mx, 2, 32));
      mx = fmaxf(mx, __shfl_xor(mx, 4, 32));
      mx = fmaxf(mx, __shfl_xor(mx, 8, 32));
      const float mnew = fmaxf(mst[r], mx);
      alpha[r] = __expf(mst[r] - mnew);
      mst[r] = mnew;
    }
    #pragma unroll
    for (int r = 0; r < 8; ++r) {
      const float p0 = __expf(s0[r] - mst[r]);
      const float p1 = __expf(s1[r] - mst[r]);
      float rs = p0 + p1;
      rs += __shfl_xor(rs, 1, 32);
      rs += __shfl_xor(rs, 2, 32);
      rs += __shfl_xor(rs, 4, 32);
      rs += __shfl_xor(rs, 8, 32);
      lst[r] = lst[r] * alpha[r] + rs;
      o0[r] *= alpha[r]; o1[r] *= alpha[r]; o2[r] *= alpha[r];
      Plds[w][r + 8 * g][col]      = (_Float16)p0;   // S C-layout -> [m][k]
      Plds[w][r + 8 * g][col + 16] = (_Float16)p1;
    }
    asm volatile("s_wait_dscnt 0" ::: "memory");      // wave-local LDS RAW
    v16h pf;                                          // P as A-fragment 16x32
    {
      const _Float16* pp = &Plds[w][col][g * 16];
      ((float4*)&pf)[0] = *(const float4*)(pp);
      ((float4*)&pf)[1] = *(const float4*)(pp + 8);
    }
    const _Float16* vbase = &Vts[cur][0][0];
    const v16h v0 = load_fragB(vbase, 32, 0,  0, lane);
    const v16h v1 = load_fragB(vbase, 32, 16, 0, lane);
    const v16h v2 = load_fragB(vbase, 32, 32, 0, lane);
    o0 = WMMA16(pf, v0, o0);
    o1 = WMMA16(pf, v1, o1);
    o2 = WMMA16(pf, v2, o2);

    if (it + 1 < 32) wait_async0();                   // tile it+1 landed
    __syncthreads();                                  // all waves done with cur
  }

  #pragma unroll
  for (int r = 0; r < 8; ++r) {
    const float inv = 1.0f / lst[r];
    const int token = b * 1024 + q0 + r + 8 * g;
    _Float16* dst = ao + (size_t)token * 384 + h * 48;
    dst[col]      = (_Float16)(o0[r] * inv);
    dst[col + 16] = (_Float16)(o1[r] * inv);
    dst[col + 32] = (_Float16)(o2[r] * inv);
  }
}

// Proj GEMM + residual, writes f32 [B,C,H,W]. Register double-buffered K loop.
__global__ __launch_bounds__(256) void proj_kernel(const _Float16* __restrict__ aoin,
                                                   const _Float16* __restrict__ w,
                                                   const float* __restrict__ x,
                                                   float* __restrict__ out) {
  const int wid  = (blockIdx.x * 256 + (int)threadIdx.x) >> 5;   // 0..3071
  const int lane = threadIdx.x & 31;
  const int mt = wid & 511;
  const int jg = wid >> 9;     // 0..5
  v8f a0 = {}, a1 = {}, a2 = {}, a3 = {};

  v16h af = load_fragA(aoin, 384, mt * 16, 0, lane);
  v16h b0 = load_fragB(w, 384, jg * 64 +  0, 0, lane);
  v16h b1 = load_fragB(w, 384, jg * 64 + 16, 0, lane);
  v16h b2 = load_fragB(w, 384, jg * 64 + 32, 0, lane);
  v16h b3 = load_fragB(w, 384, jg * 64 + 48, 0, lane);
  for (int kb = 0; kb < 384 - 32; kb += 32) {
    const int kn = kb + 32;
    const v16h afn = load_fragA(aoin, 384, mt * 16, kn, lane);
    const v16h b0n = load_fragB(w, 384, jg * 64 +  0, kn, lane);
    const v16h b1n = load_fragB(w, 384, jg * 64 + 16, kn, lane);
    const v16h b2n = load_fragB(w, 384, jg * 64 + 32, kn, lane);
    const v16h b3n = load_fragB(w, 384, jg * 64 + 48, kn, lane);
    a0 = WMMA16(af, b0, a0);
    a1 = WMMA16(af, b1, a1);
    a2 = WMMA16(af, b2, a2);
    a3 = WMMA16(af, b3, a3);
    af = afn; b0 = b0n; b1 = b1n; b2 = b2n; b3 = b3n;
  }
  a0 = WMMA16(af, b0, a0);
  a1 = WMMA16(af, b1, a1);
  a2 = WMMA16(af, b2, a2);
  a3 = WMMA16(af, b3, a3);

  const int g = lane >> 4, col = lane & 15;
  v8f accs[4] = {a0, a1, a2, a3};
  #pragma unroll
  for (int jj = 0; jj < 4; ++jj) {
    const int c = jg * 64 + jj * 16 + col;
    #pragma unroll
    for (int r = 0; r < 8; ++r) {
      const int token = mt * 16 + r + 8 * g;
      const int b = token >> 10, n = token & 1023;
      const size_t idx = ((size_t)(b * 384 + c)) * 1024 + n;
      out[idx] = x[idx] + accs[jj][r];
    }
  }
}

extern "C" void kernel_launch(void* const* d_in, const int* in_sizes, int n_in,
                              void* d_out, int out_size, void* d_ws, size_t ws_size,
                              hipStream_t stream) {
  (void)in_sizes; (void)n_in; (void)out_size; (void)ws_size;
  const float* x      = (const float*)d_in[0];
  const float* w_qkv  = (const float*)d_in[1];
  const float* w_proj = (const float*)d_in[2];
  const float* ln_w   = (const float*)d_in[3];
  const float* ln_b   = (const float*)d_in[4];
  float* out = (float*)d_out;

  char* p = (char*)d_ws;
  _Float16* tLN  = (_Float16*)p; p += (size_t)8192 * 384 * 2;        // 6.29 MB
  _Float16* w16q = (_Float16*)p; p += (size_t)1152 * 384 * 2;        // 0.88 MB
  _Float16* w16p = (_Float16*)p; p += (size_t)384 * 384 * 2;         // 0.29 MB
  _Float16* Qb   = (_Float16*)p; p += (size_t)64 * 1024 * 64 * 2;    // 8.39 MB
  _Float16* Kb   = (_Float16*)p; p += (size_t)64 * 1024 * 64 * 2;    // 8.39 MB
  _Float16* Vt   = (_Float16*)p; p += (size_t)64 * 64 * 1024 * 2;    // 8.39 MB
  _Float16* ao   = (_Float16*)p; p += (size_t)8192 * 384 * 2;        // 6.29 MB

  cvt_f16_kernel<<<(1152 * 384 + 255) / 256, 256, 0, stream>>>(w_qkv, w16q, 1152 * 384);
  cvt_f16_kernel<<<(384 * 384 + 255) / 256, 256, 0, stream>>>(w_proj, w16p, 384 * 384);
  ln_kernel<<<8192, 384, 0, stream>>>(x, ln_w, ln_b, tLN);
  {
    const int n4 = (int)(((size_t)3 * 64 * 1024 * 64 * 2) / 16);     // zero Q/K/Vt (d-pad)
    zero_kernel<<<(n4 + 255) / 256, 256, 0, stream>>>((uint4*)Qb, n4);
  }
  qkv_gemm_kernel<<<1152, 256, 0, stream>>>(tLN, w16q, Qb, Kb, Vt);
  attn_kernel<<<512, 256, 0, stream>>>(Qb, Kb, Vt, ao);
  proj_kernel<<<384, 256, 0, stream>>>(ao, w16p, x, out);
}
